// MSLongShortTermTransformer_6914897346614
// MI455X (gfx1250) — matrix operand order repaired and checked by reading, not compiled, not run on hardware
//
#include <hip/hip_runtime.h>
#include <hip/hip_bf16.h>
#include <math.h>

// ---------------------------------------------------------------------------
// CDNA5 / gfx1250 implementation of the MS long/short-term transformer block.
// wave32, WMMA f32<-f16 16x16x32 for every GEMM, fp32 accumulation.
// ---------------------------------------------------------------------------

typedef _Float16 half_t;
typedef half_t v16h __attribute__((ext_vector_type(16)));
typedef float  v8f  __attribute__((ext_vector_type(8)));
typedef float  v4f  __attribute__((ext_vector_type(4)));

#define LSEQ 1600
#define CMOD 256
#define DFF  1024
#define HW2D 40
#define NH   8
#define HD   32
#define WS15 15
#define WW   225   // 15*15
#define EPSV 1e-5f
#define QSCALE 0.17677669529663687f   // 1/sqrt(32)

// ---------------------------------------------------------------------------
// Generic WMMA GEMM:  out[M,N] = scale * (A[M,K] @ B') [+ bias] [+ add0]
//   BT == 0 : B stored as W[N,K]  (row n, col k)   -> out = A @ W^T
//   BT == 1 : B stored as V[K,N]  (row k, col n)   -> out = A @ V
// One 16x32 output tile per wave (two accumulators sharing one A fragment);
// K stepped by 32 (two v_wmma per step). M multiple of 16 at all call sites.
// OOB N columns: B index CLAMPED (branch-free loads); full-tile stores take a
// wave-uniform guard-free fast path.
// ---------------------------------------------------------------------------
template <int BT, int HAS_BIAS, int HAS_ADD>
__global__ void gemm_wmma_f16(const float* __restrict__ A, int lda,
                              const float* __restrict__ Bm, int ldb,
                              const float* __restrict__ bias,
                              const float* __restrict__ add0,
                              float* __restrict__ out, int ldo,
                              int M, int N, int K, float scale)
{
    const int lane  = threadIdx.x & 31;
    const int wave  = threadIdx.x >> 5;
    const int wpb   = blockDim.x >> 5;
    const int tilesN = (N + 31) >> 5;          // 32-wide N tiles
    const int tilesM = M >> 4;
    const int tile   = blockIdx.x * wpb + wave;
    if (tile >= tilesM * tilesN) return;

    const int mt = tile / tilesN;
    const int nt = tile % tilesN;
    const int mr = mt << 4;
    const int nb = nt << 5;

    const int hi  = lane >> 4;                 // lane half-group
    const int l15 = lane & 15;
    const int am  = mr + l15;                  // this lane's A row (M index)
    const int bn0 = nb + l15;                  // B col, first 16-wide tile
    const int bn1 = nb + 16 + l15;             // B col, second 16-wide tile
    const int bc0 = (bn0 < N) ? bn0 : (N - 1); // clamped (branch-free loads)
    const int bc1 = (bn1 < N) ? bn1 : (N - 1);

    v8f acc0 = {}, acc1 = {};
    for (int kb = 0; kb < K; kb += 32) {
        v16h a, b0, b1;
        // ---- A fragment (16x32 f16): lane holds one contiguous 16-float run
        //      K = hi*16 + [0..15]
        {
            const float* ap0 = A + am * lda + kb + hi * 16;
            const float* ap1 = A + am * lda + kb + hi * 16 + 8;
            const v4f q0 = *(const v4f*)(ap0);
            const v4f q1 = *(const v4f*)(ap0 + 4);
            const v4f q2 = *(const v4f*)(ap1);
            const v4f q3 = *(const v4f*)(ap1 + 4);
#pragma unroll
            for (int e = 0; e < 4; ++e) {
                a[e]      = (half_t)q0[e];
                a[4 + e]  = (half_t)q1[e];
                a[8 + e]  = (half_t)q2[e];
                a[12 + e] = (half_t)q3[e];
            }
        }
        // ---- B fragments (32x16 f16 each): half-VGPR slot e = K row ----
        if (BT == 0) {
            // W[N,K]: the 16 K-elements of a column's row are contiguous
            const float* bp0 = Bm + bc0 * ldb + kb + hi * 16;
            const float* bp1 = Bm + bc1 * ldb + kb + hi * 16;
            const v4f r0 = *(const v4f*)(bp0);
            const v4f r1 = *(const v4f*)(bp0 + 4);
            const v4f r2 = *(const v4f*)(bp0 + 8);
            const v4f r3 = *(const v4f*)(bp0 + 12);
            const v4f s0 = *(const v4f*)(bp1);
            const v4f s1 = *(const v4f*)(bp1 + 4);
            const v4f s2 = *(const v4f*)(bp1 + 8);
            const v4f s3 = *(const v4f*)(bp1 + 12);
#pragma unroll
            for (int e = 0; e < 4; ++e) {
                b0[e]      = (half_t)r0[e];
                b0[4 + e]  = (half_t)r1[e];
                b0[8 + e]  = (half_t)r2[e];
                b0[12 + e] = (half_t)r3[e];
                b1[e]      = (half_t)s0[e];
                b1[4 + e]  = (half_t)s1[e];
                b1[8 + e]  = (half_t)s2[e];
                b1[12 + e] = (half_t)s3[e];
            }
        } else {
            // V[K,N]: column access, strided by ldb
#pragma unroll
            for (int e = 0; e < 16; ++e) {
                const int kk = kb + hi * 16 + e;
                b0[e] = (half_t)Bm[kk * ldb + bc0];
                b1[e] = (half_t)Bm[kk * ldb + bc1];
            }
        }
        acc0 = __builtin_amdgcn_wmma_f32_16x16x32_f16(
                   false, a, false, b0, (short)0, acc0, false, false);
        acc1 = __builtin_amdgcn_wmma_f32_16x16x32_f16(
                   false, a, false, b1, (short)0, acc1, false, false);
    }

    // ---- epilogue: D VGPR r -> row (mr + hi*8 + r), col (nb[+16] + l15) ----
    const float bv0 = HAS_BIAS ? bias[bc0] : 0.0f;   // clamped idx: safe load
    const float bv1 = HAS_BIAS ? bias[bc1] : 0.0f;

    if (nb + 32 <= N) {
        // full tile: wave-uniform, completely guard-free stores
#pragma unroll
        for (int r = 0; r < 8; ++r) {
            const int row = mr + hi * 8 + r;
            float v0 = acc0[r] * scale + bv0;
            float v1 = acc1[r] * scale + bv1;
            if (HAS_ADD) {
                v0 += add0[row * ldo + bn0];
                v1 += add0[row * ldo + bn1];
            }
            out[row * ldo + bn0] = v0;
            out[row * ldo + bn1] = v1;
        }
    } else {
        // ragged tail tile (only the N=264 GEMM ever takes this path)
#pragma unroll
        for (int r = 0; r < 8; ++r) {
            const int row = mr + hi * 8 + r;
            if (bn0 < N) {
                float v0 = acc0[r] * scale + bv0;
                if (HAS_ADD) v0 += add0[row * ldo + bn0];
                out[row * ldo + bn0] = v0;
            }
            if (bn1 < N) {
                float v1 = acc1[r] * scale + bv1;
                if (HAS_ADD) v1 += add0[row * ldo + bn1];
                out[row * ldo + bn1] = v1;
            }
        }
    }
}

// ---------------------------------------------------------------------------
// LayerNorm over last dim (C=256), one block per row.
// ---------------------------------------------------------------------------
__global__ void layernorm_k(const float* __restrict__ in,
                            const float* __restrict__ g,
                            const float* __restrict__ b,
                            float* __restrict__ out, int C)
{
    const int row = blockIdx.x;
    const float* p = in + (size_t)row * C;
    __shared__ float r1[256], r2[256];
    float s = 0.f, s2 = 0.f;
    for (int c = threadIdx.x; c < C; c += blockDim.x) {
        const float x = p[c];
        s += x; s2 += x * x;
    }
    r1[threadIdx.x] = s; r2[threadIdx.x] = s2;
    __syncthreads();
    for (int st = 128; st > 0; st >>= 1) {
        if (threadIdx.x < st) {
            r1[threadIdx.x] += r1[threadIdx.x + st];
            r2[threadIdx.x] += r2[threadIdx.x + st];
        }
        __syncthreads();
    }
    const float mean = r1[0] / C;
    const float var  = r2[0] / C - mean * mean;
    const float inv  = rsqrtf(var + EPSV);
    for (int c = threadIdx.x; c < C; c += blockDim.x)
        out[(size_t)row * C + c] = (p[c] - mean) * inv * g[c] + b[c];
}

// ---------------------------------------------------------------------------
// Row softmax (in place), one block per row.
// ---------------------------------------------------------------------------
__global__ void softmax_rows_k(float* __restrict__ x, int cols)
{
    float* p = x + (size_t)blockIdx.x * cols;
    __shared__ float red[256];
    float m = -1e30f;
    for (int c = threadIdx.x; c < cols; c += blockDim.x) m = fmaxf(m, p[c]);
    red[threadIdx.x] = m; __syncthreads();
    for (int st = 128; st > 0; st >>= 1) {
        if (threadIdx.x < st) red[threadIdx.x] = fmaxf(red[threadIdx.x], red[threadIdx.x + st]);
        __syncthreads();
    }
    m = red[0]; __syncthreads();
    float sum = 0.f;
    for (int c = threadIdx.x; c < cols; c += blockDim.x) {
        const float e = __expf(p[c] - m);
        p[c] = e; sum += e;
    }
    red[threadIdx.x] = sum; __syncthreads();
    for (int st = 128; st > 0; st >>= 1) {
        if (threadIdx.x < st) red[threadIdx.x] += red[threadIdx.x + st];
        __syncthreads();
    }
    const float inv = 1.f / red[0];
    for (int c = threadIdx.x; c < cols; c += blockDim.x) p[c] *= inv;
}

// ---------------------------------------------------------------------------
// Elementwise helpers
// ---------------------------------------------------------------------------
__global__ void add2_k(const float* a, const float* b, float* o, int n)
{
    const int i = blockIdx.x * blockDim.x + threadIdx.x;
    if (i < n) o[i] = a[i] + b[i];
}
__global__ void add3_k(const float* a, const float* b, const float* c, float* o, int n)
{
    const int i = blockIdx.x * blockDim.x + threadIdx.x;
    if (i < n) o[i] = a[i] + b[i] + c[i];
}

// gK[l,c] = QV[l,c] * (1+tanh(idkv[l, c>>5])) ; gV[l,c] = QV[l,256+c] + idkv[l,8+c]
__global__ void build_gkv_k(const float* __restrict__ QV,
                            const float* __restrict__ idkv,
                            float* __restrict__ gK, float* __restrict__ gV)
{
    const int i = blockIdx.x * blockDim.x + threadIdx.x;
    if (i >= LSEQ * CMOD) return;
    const int l = i >> 8;
    const int c = i & 255;
    const float q  = QV[l * 512 + c];
    const float ik = idkv[l * 264 + (c >> 5)];
    gK[i] = q * (1.f + tanhf(ik));
    gV[i] = QV[l * 512 + 256 + c] + idkv[l * 264 + 8 + c];
}

// ---------------------------------------------------------------------------
// Local (short-term) attention: 15x15 window, dilation 1, border masking.
// One block (256 threads = 8 wave32) per pixel. Per-head softmax reduction is
// done by one wave32 per head with __shfl_xor (wave32-native).
// ---------------------------------------------------------------------------
__global__ void local_attn_k(const float* __restrict__ QV,   // [L,512], Q in [:,0:256]
                             const float* __restrict__ gK,   // [L,256]
                             const float* __restrict__ gV,   // [L,256]
                             float* __restrict__ outp)       // [L,256]
{
    const int l = blockIdx.x;
    const int y = l / HW2D, x = l % HW2D;
    const int tid = threadIdx.x;

    __shared__ float qs[CMOD];
    __shared__ float lg[NH][WW];
    __shared__ int   nbr[WW];

    qs[tid] = QV[l * 512 + tid] * QSCALE;
    if (tid < WW) {
        const int wy = tid / WS15, wx = tid % WS15;
        const int ky = y + wy - 7, kx = x + wx - 7;
        nbr[tid] = (ky >= 0 && ky < HW2D && kx >= 0 && kx < HW2D) ? (ky * HW2D + kx) : -1;
    }
    __syncthreads();

    // logits for all 8 heads x 225 taps
    for (int idx = tid; idx < NH * WW; idx += blockDim.x) {
        const int hh = idx / WW, j = idx % WW;
        const int nl = nbr[j];
        float lv = -1e30f;
        if (nl >= 0) {
            const float* kp = gK + nl * CMOD + hh * HD;
            const float* qp = qs + hh * HD;
            float s = 0.f;
#pragma unroll
            for (int d = 0; d < HD; ++d) s += qp[d] * kp[d];
            lv = s;
        }
        lg[hh][j] = lv;
    }
    __syncthreads();

    // per-head softmax: head = tid>>5 handled by one wave32
    {
        const int hh = tid >> 5, d = tid & 31;
        float m = -1e30f;
        for (int j = d; j < WW; j += 32) m = fmaxf(m, lg[hh][j]);
        for (int off = 16; off > 0; off >>= 1) m = fmaxf(m, __shfl_xor(m, off, 32));
        float sum = 0.f;
        for (int j = d; j < WW; j += 32) {
            const float lv = lg[hh][j];
            const float e = (lv > -1e29f) ? __expf(lv - m) : 0.f;
            lg[hh][j] = e; sum += e;
        }
        for (int off = 16; off > 0; off >>= 1) sum += __shfl_xor(sum, off, 32);
        const float inv = 1.f / sum;
        for (int j = d; j < WW; j += 32) lg[hh][j] *= inv;
    }
    __syncthreads();

    // weighted sum of V: thread == output channel
    {
        const int c = tid, hh = c >> 5;
        float acc = 0.f;
        for (int j = 0; j < WW; ++j) {
            const int nl = nbr[j];
            if (nl >= 0) acc += lg[hh][j] * gV[nl * CMOD + c];
        }
        outp[l * CMOD + c] = acc;
    }
}

// ---------------------------------------------------------------------------
// GroupNorm (32 groups over [32ch x 1600px]) + affine + exact GELU, in place.
// One block per group.
// ---------------------------------------------------------------------------
__global__ void groupnorm_gelu_k(float* __restrict__ x,
                                 const float* __restrict__ g,
                                 const float* __restrict__ b)
{
    const int gr = blockIdx.x;          // 0..31
    const int n  = LSEQ * 32;           // elements per group
    __shared__ float r1[256], r2[256];
    float s = 0.f, s2 = 0.f;
    for (int i = threadIdx.x; i < n; i += blockDim.x) {
        const int l = i >> 5;
        const int c = (gr << 5) + (i & 31);
        const float v = x[l * DFF + c];
        s += v; s2 += v * v;
    }
    r1[threadIdx.x] = s; r2[threadIdx.x] = s2;
    __syncthreads();
    for (int st = 128; st > 0; st >>= 1) {
        if (threadIdx.x < st) {
            r1[threadIdx.x] += r1[threadIdx.x + st];
            r2[threadIdx.x] += r2[threadIdx.x + st];
        }
        __syncthreads();
    }
    const float mean = r1[0] / n;
    const float var  = r2[0] / n - mean * mean;
    const float inv  = rsqrtf(var + EPSV);
    __syncthreads();
    for (int i = threadIdx.x; i < n; i += blockDim.x) {
        const int l = i >> 5;
        const int c = (gr << 5) + (i & 31);
        float v = (x[l * DFF + c] - mean) * inv * g[c] + b[c];
        v = 0.5f * v * (1.f + erff(v * 0.70710678118654752f));   // exact GELU
        x[l * DFF + c] = v;
    }
}

// ---------------------------------------------------------------------------
// Depthwise 5x5 conv, padding 2, channels-last [l=y*40+x, c].
// ---------------------------------------------------------------------------
__global__ void dwconv5_k(const float* __restrict__ xin,
                          const float* __restrict__ kw,     // [DFF,1,5,5]
                          float* __restrict__ xout)
{
    const int i = blockIdx.x * blockDim.x + threadIdx.x;
    if (i >= LSEQ * DFF) return;
    const int c = i & (DFF - 1);
    const int l = i >> 10;
    const int y = l / HW2D, x = l % HW2D;
    float acc = 0.f;
#pragma unroll
    for (int dy = 0; dy < 5; ++dy) {
        const int yy = y + dy - 2;
        if (yy < 0 || yy >= HW2D) continue;
#pragma unroll
        for (int dx = 0; dx < 5; ++dx) {
            const int xx = x + dx - 2;
            if (xx < 0 || xx >= HW2D) continue;
            acc += xin[(yy * HW2D + xx) * DFF + c] * kw[c * 25 + dy * 5 + dx];
        }
    }
    xout[i] = acc;
}

// ---------------------------------------------------------------------------
// Host orchestration
// ---------------------------------------------------------------------------
extern "C" void kernel_launch(void* const* d_in, const int* in_sizes, int n_in,
                              void* d_out, int out_size, void* d_ws, size_t ws_size,
                              hipStream_t stream)
{
    (void)in_sizes; (void)n_in; (void)out_size; (void)ws_size;

    const float* tgt       = (const float*)d_in[0];
    const float* id_emb    = (const float*)d_in[1];
    const float* self_pos  = (const float*)d_in[2];
    const float* ln1_g     = (const float*)d_in[3];
    const float* ln1_b     = (const float*)d_in[4];
    const float* sa_wq     = (const float*)d_in[5];
    const float* sa_bq     = (const float*)d_in[6];
    const float* sa_wk     = (const float*)d_in[7];
    const float* sa_bk     = (const float*)d_in[8];
    const float* sa_wv     = (const float*)d_in[9];
    const float* sa_bv     = (const float*)d_in[10];
    const float* sa_wo     = (const float*)d_in[11];
    const float* sa_bo     = (const float*)d_in[12];
    const float* ln2_g     = (const float*)d_in[13];
    const float* ln2_b     = (const float*)d_in[14];
    const float* w_qv      = (const float*)d_in[15];
    const float* b_qv      = (const float*)d_in[16];
    const float* w_id      = (const float*)d_in[17];
    const float* b_id      = (const float*)d_in[18];
    const float* lt_wo     = (const float*)d_in[19];
    const float* lt_bo     = (const float*)d_in[20];
    const float* st_wo     = (const float*)d_in[21];
    const float* st_bo     = (const float*)d_in[22];
    const float* ln3_g     = (const float*)d_in[23];
    const float* ln3_b     = (const float*)d_in[24];
    const float* w1        = (const float*)d_in[25];
    const float* b1        = (const float*)d_in[26];
    const float* gn_g      = (const float*)d_in[27];
    const float* gn_b      = (const float*)d_in[28];
    const float* dw_k      = (const float*)d_in[29];
    const float* w2        = (const float*)d_in[30];
    const float* b2        = (const float*)d_in[31];

    const int L = LSEQ, C = CMOD;
    const int LC = L * C;

    // ---- workspace carve-out (fp32) ----
    float* ws = (float*)d_ws;
    size_t off = 0;
    auto carve = [&](size_t nf) { float* p = ws + off; off += nf; return p; };
    float* t_ln  = carve(LC);
    float* qkin  = carve(LC);
    float* qb    = carve(LC);
    float* kb    = carve(LC);
    float* vb    = carve(LC);
    float* attn  = carve(LC);
    float* tgt1  = carve(LC);
    float* QV    = carve((size_t)L * 512);
    float* idkv  = carve((size_t)L * 264);
    float* gK    = carve(LC);
    float* gV    = carve(LC);
    float* t2    = carve(LC);
    float* t3    = carve(LC);
    float* lout  = carve(LC);
    float* tgtB  = carve(LC);
    float* x1    = carve((size_t)L * DFF);
    float* xc    = carve((size_t)L * DFF);
    float* lgt   = carve((size_t)L * L);     // per-head attention logits (10.24 MB)

    auto nblocks = [](int M, int N) {
        const int tiles = (M >> 4) * ((N + 31) >> 5);
        return (tiles + 7) / 8;
    };
    // out = scale*(A@W^T) (no bias / residual)     -- attention scores
    auto gemm_nn = [&](const float* A, int lda, const float* Bm, int ldb,
                       float* outp, int ldo, int M, int N, int K, float scale) {
        gemm_wmma_f16<0, 0, 0><<<nblocks(M, N), 256, 0, stream>>>(
            A, lda, Bm, ldb, nullptr, nullptr, outp, ldo, M, N, K, scale);
    };
    // out = A@W^T + bias                           -- projections
    auto gemm_b = [&](const float* A, int lda, const float* Bm, int ldb,
                      const float* bias, float* outp, int ldo,
                      int M, int N, int K) {
        gemm_wmma_f16<0, 1, 0><<<nblocks(M, N), 256, 0, stream>>>(
            A, lda, Bm, ldb, bias, nullptr, outp, ldo, M, N, K, 1.0f);
    };
    // out = A@W^T + bias + residual                -- output projections
    auto gemm_br = [&](const float* A, int lda, const float* Bm, int ldb,
                       const float* bias, const float* add0, float* outp, int ldo,
                       int M, int N, int K) {
        gemm_wmma_f16<0, 1, 1><<<nblocks(M, N), 256, 0, stream>>>(
            A, lda, Bm, ldb, bias, add0, outp, ldo, M, N, K, 1.0f);
    };
    // out = A@V (B stored [K,N])                   -- P @ V
    auto gemm_bt = [&](const float* A, int lda, const float* Bm, int ldb,
                       float* outp, int ldo, int M, int N, int K) {
        gemm_wmma_f16<1, 0, 0><<<nblocks(M, N), 256, 0, stream>>>(
            A, lda, Bm, ldb, nullptr, nullptr, outp, ldo, M, N, K, 1.0f);
    };

    // ================= self attention =================
    layernorm_k<<<L, 256, 0, stream>>>(tgt, ln1_g, ln1_b, t_ln, C);
    add2_k<<<(LC + 255) / 256, 256, 0, stream>>>(t_ln, self_pos, qkin, LC);

    gemm_b(qkin, C, sa_wq, C, sa_bq, qb, C, L, C, C);
    gemm_b(qkin, C, sa_wk, C, sa_bk, kb, C, L, C, C);
    gemm_b(t_ln, C, sa_wv, C, sa_bv, vb, C, L, C, C);

    for (int h = 0; h < NH; ++h) {
        gemm_nn(qb + h * HD, C, kb + h * HD, C, lgt, L, L, L, HD, QSCALE);
        softmax_rows_k<<<L, 256, 0, stream>>>(lgt, L);
        gemm_bt(lgt, L, vb + h * HD, C, attn + h * HD, C, L, HD, L);
    }
    gemm_br(attn, C, sa_wo, C, sa_bo, tgt, tgt1, C, L, C, C);

    // ================= long-term attention =================
    layernorm_k<<<L, 256, 0, stream>>>(tgt1, ln2_g, ln2_b, t_ln, C);
    gemm_b(t_ln, C, w_qv, C, b_qv, QV, 512, L, 512, C);
    gemm_b(id_emb, C, w_id, C, b_id, idkv, 264, L, 264, C);
    build_gkv_k<<<(LC + 255) / 256, 256, 0, stream>>>(QV, idkv, gK, gV);

    for (int h = 0; h < NH; ++h) {
        gemm_nn(QV + h * HD, 512, gK + h * HD, C, lgt, L, L, L, HD, QSCALE);
        softmax_rows_k<<<L, 256, 0, stream>>>(lgt, L);
        gemm_bt(lgt, L, gV + h * HD, C, attn + h * HD, C, L, HD, L);
    }
    gemm_b(attn, C, lt_wo, C, lt_bo, t2, C, L, C, C);

    // ================= short-term (local) attention =================
    local_attn_k<<<L, 256, 0, stream>>>(QV, gK, gV, lout);
    gemm_b(lout, C, st_wo, C, st_bo, t3, C, L, C, C);

    add3_k<<<(LC + 255) / 256, 256, 0, stream>>>(tgt1, t2, t3, tgtB, LC);

    // ================= FFN: w1 -> GN -> GELU -> dwconv5 -> w2 =================
    layernorm_k<<<L, 256, 0, stream>>>(tgtB, ln3_g, ln3_b, t_ln, C);
    gemm_b(t_ln, C, w1, C, b1, x1, DFF, L, DFF, C);
    groupnorm_gelu_k<<<32, 256, 0, stream>>>(x1, gn_g, gn_b);
    dwconv5_k<<<(L * DFF + 255) / 256, 256, 0, stream>>>(x1, dw_k, xc);
    gemm_br(xc, DFF, w2, DFF, b2, tgtB, (float*)d_out, C, L, C, DFF);
}
